// Grail_7550552506732
// MI455X (gfx1250) — compile-verified, matrix operand order checked by reading, not run on hardware
//
#include <hip/hip_runtime.h>
#include <hip/hip_bf16.h>
#include <math.h>

typedef __attribute__((ext_vector_type(16))) _Float16 v16h;
typedef __attribute__((ext_vector_type(8)))  _Float16 v8h;
typedef __attribute__((ext_vector_type(4)))  _Float16 v4h;
typedef __attribute__((ext_vector_type(8)))  float    v8f;
typedef __attribute__((ext_vector_type(4)))  float    v4f;

#define DN   32          // D = inp/emb dim
#define LN   3           // layers
#define NBAS 4           // num bases
#define RN   400         // aug_num_rels (rows of w_comp)
#define NTOT 50000
#define ETOT 800000
#define BGR  2000
#define NPG  25

// ---------------------------------------------------------------------------
// A-fragment K mapping (16-bit A 16x32, wave32):
//   lanes 0-15  hold row M=lane,    K = {0..7, 16..23}
//   lanes 16-31 hold row M=lane-16, K = {8..15, 24..31}
// => per lane: two contiguous 8-half chunks at half offsets base, base+16
//    with base = (lane&16) ? 8 : 0.  Each chunk = one 16-byte global_load_b128.
// B fragment mirrored (lane = column), packed by prep_weights below.
// ---------------------------------------------------------------------------

__device__ __forceinline__ v16h load_rowA_f16(const _Float16* __restrict__ row, int lane) {
    const int base = (lane & 16) ? 8 : 0;
    v8h h0 = *(const v8h*)(row + base);
    v8h h1 = *(const v8h*)(row + base + 16);
    v16h a;
#pragma unroll
    for (int i = 0; i < 8; ++i) { a[i] = h0[i]; a[i + 8] = h1[i]; }
    return a;
}

__device__ __forceinline__ v16h load_packB(const _Float16* __restrict__ pk, int blk, int lane) {
    return *(const v16h*)(pk + (size_t)blk * 512 + lane * 16);
}

// ---------------------------------------------------------------------------
// Pack f32 weights into f16 B-fragment layout.
// Sections (each "block" = 32 lanes x 16 halves = 512 halves):
//   basesPk: blk = ((l*NB + b)*2 + nh)  -> B[k][n] = bases[l][b][k][n+16*nh]
//   AwPk:    blk = ((l*4  + c)*2 + nh)  -> B[k][n] = A_w[l][n+16*nh][c*32+k]
//   slwPk:   blk = (l*2 + nh)           -> B[k][n] = slw[l][k][n+16*nh]
// ---------------------------------------------------------------------------
__global__ void prep_weights(const float* __restrict__ bases,
                             const float* __restrict__ A_w,
                             const float* __restrict__ slw,
                             _Float16* __restrict__ basesPk,
                             _Float16* __restrict__ AwPk,
                             _Float16* __restrict__ slwPk) {
    int idx = blockIdx.x * blockDim.x + threadIdx.x;   // 54*512 = 27648 threads exactly
    int blk = idx >> 9;
    int within = idx & 511;
    int lane = within >> 4;
    int i = within & 15;
    int n = lane & 15;
    int k = ((i < 8) ? i : i + 8) + ((lane >= 16) ? 8 : 0);

    if (blk < 24) {                      // bases
        int nh = blk & 1, lb = blk >> 1;
        int b = lb & 3, l = lb >> 2;
        int col = n + 16 * nh;
        basesPk[idx] = (_Float16)bases[(((size_t)l * NBAS + b) * DN + k) * DN + col];
    } else if (blk < 48) {               // A_w transposed chunks
        int q = blk - 24;
        int nh = q & 1, lc = q >> 1;
        int c = lc & 3, l = lc >> 2;
        int col = n + 16 * nh;
        AwPk[(size_t)q * 512 + within] = (_Float16)A_w[((size_t)l * DN + col) * 128 + c * 32 + k];
    } else {                             // self_loop_w
        int q = blk - 48;
        int nh = q & 1, l = q >> 1;
        int col = n + 16 * nh;
        slwPk[(size_t)q * 512 + within] = (_Float16)slw[((size_t)l * DN + k) * DN + col];
    }
}

__global__ void cvt_f32_to_f16_v4(const float* __restrict__ src,
                                  _Float16* __restrict__ dst, int n4) {
    int i = blockIdx.x * blockDim.x + threadIdx.x;
    if (i < n4) {
        v4f f = ((const v4f*)src)[i];
        v4h h;
#pragma unroll
        for (int k = 0; k < 4; ++k) h[k] = (_Float16)f[k];
        ((v4h*)dst)[i] = h;
    }
}

__global__ void zero_f32v4(float* __restrict__ p, int n4) {
    int i = blockIdx.x * blockDim.x + threadIdx.x;
    if (i < n4) ((v4f*)p)[i] = (v4f){0.f, 0.f, 0.f, 0.f};
}

// ---------------------------------------------------------------------------
// Per-edge kernel: one wave32 per 16-edge tile. All feature gathers are f16.
// ---------------------------------------------------------------------------
__global__ void __launch_bounds__(256)
edge_kernel(const _Float16* __restrict__ xh,        // [N,32] f16 features
            const int* __restrict__ esrc, const int* __restrict__ edst,
            const int* __restrict__ etype, const int* __restrict__ elabel,
            const float* __restrict__ w_comp_l,      // [R, NB]
            const _Float16* __restrict__ attnH,      // [NR,32] f16
            const _Float16* __restrict__ basesPk_l,  // 8 blocks
            const _Float16* __restrict__ AwPk_l,     // 8 blocks
            const float* __restrict__ Ab_l,          // [32]
            const float* __restrict__ Bw_l,          // [32]
            const float* __restrict__ Bb_l,          // [1]
            float* __restrict__ agg) {
    const int lane  = threadIdx.x & 31;
    const int tile  = blockIdx.x * (blockDim.x >> 5) + (threadIdx.x >> 5);
    const int ebase = tile * 16;
    const int r     = lane & 15;
    const int e     = ebase + r;

    const int src = esrc[e];
    const int dst = edst[e];
    const int et  = etype[e];
    const int el  = elabel[e];

    // A fragments (16x32 f16 each) -- direct f16 gathers, no conversion
    v16h a_xs = load_rowA_f16(xh + (size_t)src * DN, lane);
    v16h a_xd = load_rowA_f16(xh + (size_t)dst * DN, lane);
    v16h a_t  = load_rowA_f16(attnH + (size_t)et * DN, lane);
    v16h a_l  = load_rowA_f16(attnH + (size_t)el * DN, lane);

    // msg = sum_b (coef_b .rows xs) @ bases_b  -- coef folded into A rows
    v4f coef = *(const v4f*)(w_comp_l + (size_t)et * NBAS);
    v8f m0 = {}, m1 = {};
#pragma unroll
    for (int b = 0; b < NBAS; ++b) {
        _Float16 cs = (_Float16)coef[b];
        v16h as;
#pragma unroll
        for (int i = 0; i < 16; ++i) as[i] = a_xs[i] * cs;
        v16h bb0 = load_packB(basesPk_l, b * 2 + 0, lane);
        v16h bb1 = load_packB(basesPk_l, b * 2 + 1, lane);
        m0 = __builtin_amdgcn_wmma_f32_16x16x32_f16(false, as, false, bb0, (short)0, m0, false, false);
        m1 = __builtin_amdgcn_wmma_f32_16x16x32_f16(false, as, false, bb1, (short)0, m1, false, false);
    }

    // attention pre-activation: [xs,xd,attn_t,attn_l] (16x128) @ A_w^T (128x32)
    v8f t0 = {}, t1 = {};
    v16h achunk[4] = {a_xs, a_xd, a_t, a_l};
#pragma unroll
    for (int c = 0; c < 4; ++c) {
        v16h wb0 = load_packB(AwPk_l, c * 2 + 0, lane);
        v16h wb1 = load_packB(AwPk_l, c * 2 + 1, lane);
        t0 = __builtin_amdgcn_wmma_f32_16x16x32_f16(false, achunk[c], false, wb0, (short)0, t0, false, false);
        t1 = __builtin_amdgcn_wmma_f32_16x16x32_f16(false, achunk[c], false, wb1, (short)0, t1, false, false);
    }

    // alpha_r = sigmoid( relu(t + Ab) . Bw + Bb ) ; C layout: lane=col, vgpr j=row
    const int col = lane & 15;
    const float ab0 = Ab_l[col], ab1 = Ab_l[col + 16];
    const float bw0 = Bw_l[col], bw1 = Bw_l[col + 16];
    const float bb  = Bb_l[0];
    float alphas[8];
#pragma unroll
    for (int j = 0; j < 8; ++j) {
        float s = fmaxf(t0[j] + ab0, 0.f) * bw0 + fmaxf(t1[j] + ab1, 0.f) * bw1;
        s += __shfl_xor(s, 1, 16);
        s += __shfl_xor(s, 2, 16);
        s += __shfl_xor(s, 4, 16);
        s += __shfl_xor(s, 8, 16);
        alphas[j] = 1.f / (1.f + __expf(-(s + bb)));
    }

    // scatter: rows serviced by this lane group = ebase + 8*(lane>>4) + j
    const int rowbase = ebase + ((lane >> 4) << 3);
#pragma unroll
    for (int j = 0; j < 8; ++j) {
        int d = edst[rowbase + j];
        float a = alphas[j];
        atomicAdd(&agg[(size_t)d * DN + col],      a * m0[j]);
        atomicAdd(&agg[(size_t)d * DN + col + 16], a * m1[j]);
    }
}

// ---------------------------------------------------------------------------
// Node update: x_new = relu(x @ self_loop_w + agg).
// Writes f32 reprs (for readout) and f16 x (for next layer's gathers).
// ---------------------------------------------------------------------------
__global__ void __launch_bounds__(160)
node_kernel(const _Float16* __restrict__ xh_in,
            const _Float16* __restrict__ slwPk_l,
            const float* __restrict__ agg,
            float* __restrict__ reprs,
            _Float16* __restrict__ xh_out, int lidx) {
    const int lane  = threadIdx.x & 31;
    const int tile  = blockIdx.x * (blockDim.x >> 5) + (threadIdx.x >> 5);
    const int nbase = tile * 16;
    const int r     = lane & 15;

    v16h a  = load_rowA_f16(xh_in + (size_t)(nbase + r) * DN, lane);
    v16h b0 = load_packB(slwPk_l, 0, lane);
    v16h b1 = load_packB(slwPk_l, 1, lane);

    const int col = lane & 15;
    const int rj  = nbase + ((lane >> 4) << 3);
    v8f c0, c1;
#pragma unroll
    for (int j = 0; j < 8; ++j) {
        c0[j] = agg[(size_t)(rj + j) * DN + col];
        c1[j] = agg[(size_t)(rj + j) * DN + col + 16];
    }
    c0 = __builtin_amdgcn_wmma_f32_16x16x32_f16(false, a, false, b0, (short)0, c0, false, false);
    c1 = __builtin_amdgcn_wmma_f32_16x16x32_f16(false, a, false, b1, (short)0, c1, false, false);
#pragma unroll
    for (int j = 0; j < 8; ++j) {
        float v0 = fmaxf(c0[j], 0.f);
        float v1 = fmaxf(c1[j], 0.f);
        reprs[(size_t)(rj + j) * (LN * DN) + lidx * DN + col]      = v0;
        reprs[(size_t)(rj + j) * (LN * DN) + lidx * DN + col + 16] = v1;
        xh_out[(size_t)(rj + j) * DN + col]      = (_Float16)v0;
        xh_out[(size_t)(rj + j) * DN + col + 16] = (_Float16)v1;
    }
}

// ---------------------------------------------------------------------------
// Readout: one wave per graph (nodes are contiguous, NP=25 per graph).
// ---------------------------------------------------------------------------
__global__ void __launch_bounds__(256)
readout_kernel(const float* __restrict__ reprs,
               const int* __restrict__ head_ids, const int* __restrict__ tail_ids,
               const int* __restrict__ rel_labels, const float* __restrict__ rel_tab,
               const float* __restrict__ fc_w, const float* __restrict__ fc_b,
               float* __restrict__ out) {
    const int lane = threadIdx.x & 31;
    const int g    = blockIdx.x * (blockDim.x >> 5) + (threadIdx.x >> 5);

    float m0 = 0.f, m1 = 0.f, m2 = 0.f;
    const int nbase = g * NPG;
    for (int i = 0; i < NPG; ++i) {
        const float* row = reprs + (size_t)(nbase + i) * (LN * DN);
        m0 += row[lane]; m1 += row[lane + 32]; m2 += row[lane + 64];
    }
    const float inv = 1.f / (float)NPG;
    const float* hrow = reprs + (size_t)head_ids[g] * (LN * DN);
    const float* trow = reprs + (size_t)tail_ids[g] * (LN * DN);
    const int rl = rel_labels[g];

    float s = m0 * inv * fc_w[lane] + m1 * inv * fc_w[lane + 32] + m2 * inv * fc_w[lane + 64];
    s += hrow[lane] * fc_w[96 + lane] + hrow[lane + 32] * fc_w[96 + lane + 32] + hrow[lane + 64] * fc_w[96 + lane + 64];
    s += trow[lane] * fc_w[192 + lane] + trow[lane + 32] * fc_w[192 + lane + 32] + trow[lane + 64] * fc_w[192 + lane + 64];
    s += rel_tab[(size_t)rl * DN + lane] * fc_w[288 + lane];

    s += __shfl_xor(s, 1, 32);
    s += __shfl_xor(s, 2, 32);
    s += __shfl_xor(s, 4, 32);
    s += __shfl_xor(s, 8, 32);
    s += __shfl_xor(s, 16, 32);
    if (lane == 0) out[g] = s + fc_b[0];
}

// ---------------------------------------------------------------------------
extern "C" void kernel_launch(void* const* d_in, const int* in_sizes, int n_in,
                              void* d_out, int out_size, void* d_ws, size_t ws_size,
                              hipStream_t stream) {
    const float* feat       = (const float*)d_in[0];
    const int*   edge_src   = (const int*)d_in[1];
    const int*   edge_dst   = (const int*)d_in[2];
    const int*   edge_type  = (const int*)d_in[3];
    const int*   edge_label = (const int*)d_in[4];
    // d_in[5] = graph_ids (contiguous by construction, unused)
    const int*   head_ids   = (const int*)d_in[6];
    const int*   tail_ids   = (const int*)d_in[7];
    const int*   rel_labels = (const int*)d_in[8];
    const float* bases      = (const float*)d_in[9];
    const float* w_comp     = (const float*)d_in[10];
    const float* slw        = (const float*)d_in[11];
    const float* A_w        = (const float*)d_in[12];
    const float* A_b        = (const float*)d_in[13];
    const float* B_w        = (const float*)d_in[14];
    const float* B_b        = (const float*)d_in[15];
    const float* attn_tab   = (const float*)d_in[16];
    const float* rel_tab    = (const float*)d_in[17];
    const float* fc_w       = (const float*)d_in[18];
    const float* fc_b       = (const float*)d_in[19];
    float* out = (float*)d_out;

    char* ws = (char*)d_ws;
    float*     agg     = (float*)ws;                    //  6,400,000 B
    float*     reprs   = (float*)(ws + 6400000);        // 19,200,000 B
    _Float16*  featH   = (_Float16*)(ws + 25600000);    //  3,200,000 B
    _Float16*  xhA     = (_Float16*)(ws + 28800000);    //  3,200,000 B
    _Float16*  xhB     = (_Float16*)(ws + 32000000);    //  3,200,000 B
    _Float16*  attnH   = (_Float16*)(ws + 35200000);    //     12,800 B
    _Float16*  basesPk = (_Float16*)(ws + 35212800);    //     24,576 B
    _Float16*  AwPk    = (_Float16*)(ws + 35237376);    //     24,576 B
    _Float16*  slwPk   = (_Float16*)(ws + 35261952);    //      6,144 B

    // 1) pack weights to f16 B-fragments (54*512 = 27648 threads exactly)
    prep_weights<<<108, 256, 0, stream>>>(bases, A_w, slw, basesPk, AwPk, slwPk);

    // 2) f16 caches of feat (N*32 = 1.6M -> 400K v4) and attn table (6400 -> 1600 v4)
    cvt_f32_to_f16_v4<<<(400000 + 255) / 256, 256, 0, stream>>>(feat, featH, 400000);
    cvt_f32_to_f16_v4<<<(1600 + 255) / 256, 256, 0, stream>>>(attn_tab, attnH, 1600);

    const _Float16* xh_in = featH;
    _Float16* xh_pp[2] = {xhA, xhB};

    for (int l = 0; l < LN; ++l) {
        // 3) zero aggregation buffer (N*D = 1.6M floats = 400K float4)
        zero_f32v4<<<(400000 + 255) / 256, 256, 0, stream>>>(agg, 400000);

        // 4) per-edge: 800000/16 = 50000 tiles, 8 waves/block -> 6250 blocks
        edge_kernel<<<6250, 256, 0, stream>>>(
            xh_in, edge_src, edge_dst, edge_type, edge_label,
            w_comp + (size_t)l * RN * NBAS, attnH,
            basesPk + (size_t)l * 8 * 512, AwPk + (size_t)l * 8 * 512,
            A_b + (size_t)l * DN, B_w + (size_t)l * DN, B_b + l, agg);

        // 5) node update: 50000/16 = 3125 tiles, 5 waves/block -> 625 blocks
        _Float16* xh_out = xh_pp[l & 1];
        node_kernel<<<625, 160, 0, stream>>>(
            xh_in, slwPk + (size_t)l * 2 * 512, agg, reprs, xh_out, l);
        xh_in = xh_out;
    }

    // 6) readout: 2000 graphs, 8 waves/block -> 250 blocks
    readout_kernel<<<250, 256, 0, stream>>>(
        reprs, head_ids, tail_ids, rel_labels, rel_tab, fc_w, fc_b, out);
}
